// MoELayer_tp_6846177870127
// MI455X (gfx1250) — matrix operand order repaired and checked by reading, not compiled, not run on hardware
//
// MoE transformer layer for MI455X (gfx1250), wave32 + WMMA bf16.
// Roofline: ~137 GFLOP (dense experts) vs ~150 MB HBM (~6.5 us @ 23.3 TB/s)
// => matrix-unit bound; all GEMMs use v_wmma_f32_16x16x32_bf16.
// GEMM: 8 waves/block, each wave computes a 16x64 strip (4 WMMA accumulators,
// A-fragment reused 4x) -> block tile 128x64, LDS-staged (8KB A + 4KB B).
#include <hip/hip_runtime.h>
#include <math.h>

// ---------------- problem dims ----------------
#define SDIM 1024
#define BDIM 2
#define HDIM 1024
#define NHD  16
#define HDD  64
#define EDIM 8
#define FDIM 2048
#define TDIM (SDIM*BDIM)          // 2048 tokens
#define EPSV 1e-6f

typedef __attribute__((ext_vector_type(16))) __bf16 v16bf;
typedef __attribute__((ext_vector_type(8)))  float  v8f;
typedef __attribute__((ext_vector_type(4)))  unsigned int u32x4;
typedef __attribute__((ext_vector_type(8)))  int    i32x8;
typedef __attribute__((ext_vector_type(4)))  int    i32x4;

union FragBf { v16bf v; uint4 q[2]; };

static __device__ __forceinline__ v8f wmma_bf16(const FragBf& a, const FragBf& b, v8f c) {
#if defined(__gfx1250__)
  // 8 args: (neg_a, A, neg_b, B, c_mod, C, reuse_a, reuse_b)
  return __builtin_amdgcn_wmma_f32_16x16x32_bf16(false, a.v, false, b.v, (short)0, c, false, false);
#else
  return c;
#endif
}

// ---------------- CDNA5 TDM probe: NULL-descriptor tensor load (no-op) ----------------
__global__ void cdna5_tdm_probe_kernel() {
#if defined(__gfx1250__) && __has_builtin(__builtin_amdgcn_tensor_load_to_lds)
  __shared__ long lds_sink[16];
  u32x4 g0 = {};                 // group0.count==0 -> NULL tensor descriptor (no transfer)
  i32x8 g1 = {};
  i32x4 g2 = {}, g3 = {};
  i32x8 g4 = {};
  // clang-23 / therock-10.0 arity: (g0 u32x4, g1 i32x8, i32x4, i32x4, i32x8, cpol)
  __builtin_amdgcn_tensor_load_to_lds(g0, g1, g2, g3, g4, 0);
#if __has_builtin(__builtin_amdgcn_s_wait_tensorcnt)
  __builtin_amdgcn_s_wait_tensorcnt(0);
#endif
  if (threadIdx.x == 31) lds_sink[0] = 0;
  (void)lds_sink;
#endif
}

// ---------------- weight prep ----------------
__global__ __launch_bounds__(256) void cast_bf16_kernel(const float* __restrict__ src,
                                                        __bf16* __restrict__ dst, long n) {
  for (long i = (long)blockIdx.x * blockDim.x + threadIdx.x; i < n;
       i += (long)gridDim.x * blockDim.x)
    dst[i] = (__bf16)src[i];
}

// dst[c*rows + r] = (bf16)src[r*cols + c]  (per-expert batch in blockIdx.z)
__global__ __launch_bounds__(256) void transpose_cast_kernel(const float* __restrict__ src,
                                                             __bf16* __restrict__ dst,
                                                             int rows, int cols) {
  __shared__ float tile[32][33];
  const long eoff = (long)blockIdx.z * rows * cols;
  src += eoff; dst += eoff;
  const int c0 = blockIdx.x * 32, r0 = blockIdx.y * 32;
  const int tx = threadIdx.x, ty = threadIdx.y;
#pragma unroll
  for (int i = 0; i < 32; i += 8)
    tile[ty + i][tx] = src[(long)(r0 + ty + i) * cols + (c0 + tx)];
  __syncthreads();
#pragma unroll
  for (int i = 0; i < 32; i += 8)
    dst[(long)(c0 + ty + i) * rows + (r0 + tx)] = (__bf16)tile[tx][ty + i];
}

// ---------------- fused RMSNorm + bf16 cast (one token per block) ----------------
__global__ __launch_bounds__(256) void rmsnorm_cast_kernel(const float* __restrict__ x,
                                                           const float* __restrict__ w,
                                                           __bf16* __restrict__ out) {
  const int t = blockIdx.x;
  const float4 v = ((const float4*)(x + (long)t * HDIM))[threadIdx.x];
  float ss = v.x*v.x + v.y*v.y + v.z*v.z + v.w*v.w;
#pragma unroll
  for (int off = 16; off >= 1; off >>= 1) ss += __shfl_xor(ss, off, 32);
  __shared__ float wsum[8];
  if ((threadIdx.x & 31) == 0) wsum[threadIdx.x >> 5] = ss;
  __syncthreads();
  float tot = 0.f;
#pragma unroll
  for (int i = 0; i < 8; ++i) tot += wsum[i];
  const float r = rsqrtf(tot * (1.0f / HDIM) + EPSV);
  const float4 wv = ((const float4*)w)[threadIdx.x];
  const long o = (long)t * HDIM + threadIdx.x * 4;
  out[o + 0] = (__bf16)(v.x * r * wv.x);
  out[o + 1] = (__bf16)(v.y * r * wv.y);
  out[o + 2] = (__bf16)(v.z * r * wv.z);
  out[o + 3] = (__bf16)(v.w * r * wv.w);
}

// ---------------- WMMA bf16 NT GEMM: C[M,N] = A[M,K] * B[N,K]^T ----------------
// Block tile 128x64, 8 waves; wave w owns rows [w*16, w*16+16), all 64 cols:
// per K-step: 1 A-frag (reused 4x) + 4 B-frags -> 4 WMMAs.
__global__ __launch_bounds__(256) void gemm_bf16_nt(const __bf16* __restrict__ A,
                                                    const __bf16* __restrict__ B,
                                                    float* __restrict__ C,
                                                    int M, int N, int K) {
  __shared__ __bf16 sA[128][32];   // 8 KB
  __shared__ __bf16 sB[64][32];    // 4 KB
  const int tid  = threadIdx.x;
  const int wave = tid >> 5;
  const int lane = tid & 31;
  const int half = lane >> 4;
  const int l16  = lane & 15;
  const int mi = wave << 4;                       // wave's 16-row strip
  const long mblk = (long)blockIdx.y << 7;        // 128 rows / block
  const long nblk = (long)blockIdx.x << 6;        // 64 cols / block

  const int ar = tid >> 1, ac = (tid & 1) << 4;   // A stage: 16 bf16 / thread
  const int br = tid >> 2, bc = (tid & 3) << 3;   // B stage: 8 bf16 / thread

  v8f acc[4] = {};
  for (int kb = 0; kb < K; kb += 32) {
    const __bf16* ga = A + (mblk + ar) * (long)K + kb + ac;
    const __bf16* gb = B + (nblk + br) * (long)K + kb + bc;
    *(uint4*)&sA[ar][ac]     = *(const uint4*)ga;
    *(uint4*)&sA[ar][ac + 8] = *(const uint4*)(ga + 8);
    *(uint4*)&sB[br][bc]     = *(const uint4*)gb;
    if (kb + 32 < K) {                      // hint next K tile -> global_prefetch
      __builtin_prefetch(ga + 32, 0, 1);
      __builtin_prefetch(gb + 32, 0, 1);
    }
    __syncthreads();
    FragBf fa;
    const __bf16* pa = &sA[mi + l16][half << 3];  // A frag: K = h*8..+7, 16+h*8..+7
    fa.q[0] = *(const uint4*)pa;
    fa.q[1] = *(const uint4*)(pa + 16);
#pragma unroll
    for (int j = 0; j < 4; ++j) {                 // 4 N-subtiles, A reused
      FragBf fb;
      const __bf16* pb = &sB[(j << 4) + l16][half << 3];  // B frag: K = h*8..+7, 16+h*8..+7
      fb.q[0] = *(const uint4*)pb;
      fb.q[1] = *(const uint4*)(pb + 16);
      acc[j] = wmma_bf16(fa, fb, acc[j]);
    }
    __syncthreads();
  }
  const long row0 = mblk + mi + (half << 3);      // C row = vgpr_i + 8*(lane>=16)
#pragma unroll
  for (int j = 0; j < 4; ++j) {
    const long col = nblk + (j << 4) + l16;
#pragma unroll
    for (int i = 0; i < 8; ++i)
      C[(row0 + i) * (long)N + col] = acc[j][i];
  }
}

// ---------------- RoPE + pack q/k row-major, v transposed [HD,S]; fold 1/sqrt(HD) into q ----
__global__ __launch_bounds__(64) void rope_pack_kernel(const float* __restrict__ qkv,
                                                       __bf16* __restrict__ qb,
                                                       __bf16* __restrict__ kb,
                                                       __bf16* __restrict__ vt) {
  const int s  = blockIdx.x;
  const int bh = blockIdx.y;              // b*NH + nh
  const int b  = bh / NHD, nh = bh % NHD;
  const int d  = threadIdx.x;
  const long t = (long)s * BDIM + b;
  const float* base = qkv + t * (3 * HDIM) + nh * (3 * HDD);
  const float q = base[d];
  const float k = base[HDD + d];
  const float v = base[2 * HDD + d];
  const int   i   = d & 31;
  const float inv = __powf(10000.0f, -(float)i * (1.0f / 32.0f));
  const float ang = (float)s * inv;
  const float c = __cosf(ang), sn = __sinf(ang);
  const float rq = (d < 32) ? -base[d + 32]       : base[d - 32];
  const float rk = (d < 32) ? -base[HDD + d + 32] : base[HDD + d - 32];
  const long idx = ((long)bh * SDIM + s) * HDD + d;
  qb[idx] = (__bf16)((q * c + rq * sn) * 0.125f);   // 1/sqrt(64)
  kb[idx] = (__bf16)(k * c + rk * sn);
  vt[((long)bh * HDD + d) * SDIM + s] = (__bf16)v;
}

// ---------------- flash attention: 4 waves/WG, 16 q-rows/wave, online softmax ----------------
__global__ __launch_bounds__(128) void flash_attn_kernel(const __bf16* __restrict__ qb,
                                                         const __bf16* __restrict__ kbm,
                                                         const __bf16* __restrict__ vt,
                                                         __bf16* __restrict__ ctx) {
  const int bh   = blockIdx.y;
  const int wave = threadIdx.x >> 5;
  const int lane = threadIdx.x & 31;
  const int half = lane >> 4;
  const int l16  = lane & 15;
  const int qbase = blockIdx.x * 64 + wave * 16;

  const __bf16* Q  = qb  + (long)bh * SDIM * HDD;
  const __bf16* Km = kbm + (long)bh * SDIM * HDD;
  const __bf16* Vm = vt  + (long)bh * HDD * SDIM;

  FragBf qf0, qf1;                       // Q A-frags for d=0..31 and d=32..63
  {
    const __bf16* qp = Q + (long)(qbase + l16) * HDD + (half << 3);
    qf0.q[0] = *(const uint4*)qp;
    qf0.q[1] = *(const uint4*)(qp + 16);
    qf1.q[0] = *(const uint4*)(qp + 32);
    qf1.q[1] = *(const uint4*)(qp + 48);
  }

  float m_i[8], l_i[8];
  v8f cacc[4] = {};
#pragma unroll
  for (int i = 0; i < 8; ++i) { m_i[i] = -3.0e30f; l_i[i] = 0.f; }

  __shared__ __bf16 psh[4][16][32];      // per-wave P tile staging

  const int kend = qbase + 16;           // causal: cols <= last row of this wave
  for (int kb0 = 0; kb0 < kend; kb0 += 32) {
    v8f s0 = {}, s1 = {};
    {
      FragBf f;
      const __bf16* p0 = Km + (long)(kb0 + l16) * HDD + (half << 3);
      f.q[0] = *(const uint4*)p0;        f.q[1] = *(const uint4*)(p0 + 16);
      s0 = wmma_bf16(qf0, f, s0);
      f.q[0] = *(const uint4*)(p0 + 32); f.q[1] = *(const uint4*)(p0 + 48);
      s0 = wmma_bf16(qf1, f, s0);
      const __bf16* p1 = Km + (long)(kb0 + 16 + l16) * HDD + (half << 3);
      f.q[0] = *(const uint4*)p1;        f.q[1] = *(const uint4*)(p1 + 16);
      s1 = wmma_bf16(qf0, f, s1);
      f.q[0] = *(const uint4*)(p1 + 32); f.q[1] = *(const uint4*)(p1 + 48);
      s1 = wmma_bf16(qf1, f, s1);
    }
    float mt[8];
#pragma unroll
    for (int i = 0; i < 8; ++i) {        // causal mask (cndmask, EXEC untouched)
      const int r = qbase + i + (half << 3);
      if (kb0 + l16      > r) s0[i] = -3.0e30f;
      if (kb0 + 16 + l16 > r) s1[i] = -3.0e30f;
      mt[i] = fmaxf(s0[i], s1[i]);
    }
#pragma unroll
    for (int off = 1; off < 16; off <<= 1)
#pragma unroll
      for (int i = 0; i < 8; ++i)
        mt[i] = fmaxf(mt[i], __shfl_xor(mt[i], off, 32));
    float rs[8];
#pragma unroll
    for (int i = 0; i < 8; ++i) {
      const float mnew = fmaxf(m_i[i], mt[i]);
      const float sc = __expf(m_i[i] - mnew);
      const float p0 = __expf(s0[i] - mnew);
      const float p1 = __expf(s1[i] - mnew);
      s0[i] = p0; s1[i] = p1;
      rs[i] = p0 + p1;
      l_i[i] *= sc;
      m_i[i] = mnew;
#pragma unroll
      for (int j = 0; j < 4; ++j) cacc[j][i] *= sc;
    }
#pragma unroll
    for (int off = 1; off < 16; off <<= 1)
#pragma unroll
      for (int i = 0; i < 8; ++i) rs[i] += __shfl_xor(rs[i], off, 32);
#pragma unroll
    for (int i = 0; i < 8; ++i) l_i[i] += rs[i];

    // P (C layout) -> LDS -> A-fragment layout; wave-local, no block barrier needed
#pragma unroll
    for (int i = 0; i < 8; ++i) {
      psh[wave][i + (half << 3)][l16]      = (__bf16)s0[i];
      psh[wave][i + (half << 3)][l16 + 16] = (__bf16)s1[i];
    }
#if defined(__gfx1250__)
    asm volatile("s_wait_dscnt 0" ::: "memory");
#endif
    FragBf pf;
    const __bf16* pp = &psh[wave][l16][half << 3];
    pf.q[0] = *(const uint4*)pp;
    pf.q[1] = *(const uint4*)(pp + 16);

#pragma unroll
    for (int j = 0; j < 4; ++j) {        // ctx += P(16x32) * V(32x16) per d-tile
      FragBf vf;
      const __bf16* pv = Vm + (long)(j * 16 + l16) * SDIM + kb0 + (half << 3);
      vf.q[0] = *(const uint4*)pv;
      vf.q[1] = *(const uint4*)(pv + 16);
      cacc[j] = wmma_bf16(pf, vf, cacc[j]);
    }
  }

  const int b = bh / NHD, nh = bh % NHD;
#pragma unroll
  for (int i = 0; i < 8; ++i) {
    const int s = qbase + i + (half << 3);
    const float inv = 1.0f / l_i[i];
    const long base = ((long)s * BDIM + b) * HDIM + nh * HDD;
#pragma unroll
    for (int j = 0; j < 4; ++j)
      ctx[base + j * 16 + l16] = (__bf16)(cacc[j][i] * inv);
  }
}

// ---------------- router ----------------
__global__ __launch_bounds__(256) void router_logits_kernel(const __bf16* __restrict__ tok,
                                                            const float* __restrict__ rw,
                                                            float* __restrict__ logits) {
  const int t = blockIdx.x;
  const int wave = threadIdx.x >> 5, lane = threadIdx.x & 31;   // wave == expert
  const __bf16* x = tok + (long)t * HDIM;
  const float* wr = rw + (long)wave * HDIM;
  float s = 0.f;
  for (int k = lane; k < HDIM; k += 32) s += (float)x[k] * wr[k];
#pragma unroll
  for (int off = 16; off >= 1; off >>= 1) s += __shfl_xor(s, off, 32);
  if (lane == 0) logits[t * EDIM + wave] = s;
}

__global__ __launch_bounds__(256) void router_topk_kernel(const float* __restrict__ logits,
                                                          float* __restrict__ probs) {
  const int t = blockIdx.x * blockDim.x + threadIdx.x;
  if (t >= TDIM) return;
  float l[EDIM];
  float mx = -3.0e30f;
#pragma unroll
  for (int e = 0; e < EDIM; ++e) { l[e] = logits[t * EDIM + e]; mx = fmaxf(mx, l[e]); }
  float sum = 0.f;
#pragma unroll
  for (int e = 0; e < EDIM; ++e) { l[e] = __expf(l[e] - mx); sum += l[e]; }
  const float inv = 1.0f / sum;
#pragma unroll
  for (int e = 0; e < EDIM; ++e) l[e] *= inv;
  int i1 = 0;
#pragma unroll
  for (int e = 1; e < EDIM; ++e) if (l[e] > l[i1]) i1 = e;
  int i2 = (i1 == 0) ? 1 : 0;
#pragma unroll
  for (int e = 0; e < EDIM; ++e) if (e != i1 && l[e] > l[i2]) i2 = e;
#pragma unroll
  for (int e = 0; e < EDIM; ++e)
    probs[t * EDIM + e] = (e == i1 || e == i2) ? l[e] : 0.f;
}

// ---------------- elementwise ----------------
__global__ __launch_bounds__(256) void add_inplace_kernel(float* __restrict__ a,
                                                          const float* __restrict__ b, long n) {
  for (long i = (long)blockIdx.x * blockDim.x + threadIdx.x; i < n;
       i += (long)gridDim.x * blockDim.x) a[i] += b[i];
}
__global__ __launch_bounds__(256) void zero_kernel(float* __restrict__ p, long n) {
  for (long i = (long)blockIdx.x * blockDim.x + threadIdx.x; i < n;
       i += (long)gridDim.x * blockDim.x) p[i] = 0.f;
}
__global__ __launch_bounds__(256) void gelu_cast_kernel(const float* __restrict__ x,
                                                        __bf16* __restrict__ y, long n) {
  for (long i = (long)blockIdx.x * blockDim.x + threadIdx.x; i < n;
       i += (long)gridDim.x * blockDim.x) {
    const float v = x[i];
    y[i] = (__bf16)(0.5f * v * (1.0f + erff(v * 0.70710678f)));   // exact GELU
  }
}
__global__ __launch_bounds__(256) void combine_kernel(float* __restrict__ yacc,
                                                      const float* __restrict__ y,
                                                      const float* __restrict__ probs,
                                                      int e, long n) {
  for (long i = (long)blockIdx.x * blockDim.x + threadIdx.x; i < n;
       i += (long)gridDim.x * blockDim.x) {
    const long t = i >> 10;                 // HDIM == 1024
    yacc[i] += probs[t * EDIM + e] * y[i];
  }
}
__global__ __launch_bounds__(256) void final_add_kernel(const float* __restrict__ yacc,
                                                        const float* __restrict__ hidden,
                                                        float* __restrict__ out, long n) {
  for (long i = (long)blockIdx.x * blockDim.x + threadIdx.x; i < n;
       i += (long)gridDim.x * blockDim.x) out[i] = yacc[i] + hidden[i];
}

// ---------------- host orchestration ----------------
extern "C" void kernel_launch(void* const* d_in, const int* in_sizes, int n_in,
                              void* d_out, int out_size, void* d_ws, size_t ws_size,
                              hipStream_t stream) {
  (void)in_sizes; (void)n_in; (void)out_size; (void)ws_size;
  const float* hs    = (const float*)d_in[0];
  const float* ln1w  = (const float*)d_in[1];
  const float* ln2w  = (const float*)d_in[2];
  const float* wqkv  = (const float*)d_in[3];
  const float* wproj = (const float*)d_in[4];
  const float* rw    = (const float*)d_in[5];
  const float* w1    = (const float*)d_in[6];
  const float* w2    = (const float*)d_in[7];
  float* out = (float*)d_out;

  char* ws = (char*)d_ws;
  size_t off = 0;
  auto alloc = [&](size_t bytes) { char* p = ws + off; off += (bytes + 255) & ~(size_t)255; return p; };
  __bf16* wqkv_b  = (__bf16*)alloc((size_t)3 * HDIM * HDIM * 2);
  __bf16* wproj_b = (__bf16*)alloc((size_t)HDIM * HDIM * 2);
  __bf16* w1t_b   = (__bf16*)alloc((size_t)EDIM * HDIM * FDIM * 2);  // [E,F,H]
  __bf16* w2t_b   = (__bf16*)alloc((size_t)EDIM * HDIM * FDIM * 2);  // [E,H,F]
  __bf16* xb      = (__bf16*)alloc((size_t)TDIM * HDIM * 2);         // ln1 out, reused for ln2 out
  char*   big0    = alloc((size_t)TDIM * 3 * HDIM * 4);              // qkv_f | h1_f + h1g_b
  char*   big1    = alloc((size_t)3 * TDIM * HDIM * 2);              // q,k,vt | y_f
  __bf16* ctx_b   = (__bf16*)alloc((size_t)TDIM * HDIM * 2);
  float*  hidden  = (float*)alloc((size_t)TDIM * HDIM * 4);
  float*  yacc    = (float*)alloc((size_t)TDIM * HDIM * 4);
  float*  logits  = (float*)alloc((size_t)TDIM * EDIM * 4);
  float*  probs   = (float*)alloc((size_t)TDIM * EDIM * 4);

  float*  qkv_f = (float*)big0;
  float*  h1_f  = (float*)big0;
  __bf16* h1g_b = (__bf16*)(big0 + (size_t)TDIM * FDIM * 4);
  __bf16* qb    = (__bf16*)big1;
  __bf16* kbuf  = qb + (size_t)TDIM * HDIM;
  __bf16* vtb   = kbuf + (size_t)TDIM * HDIM;
  float*  y_f   = (float*)big1;

  // CDNA5 feature probe (TDM null descriptor + tensorcnt wait)
  cdna5_tdm_probe_kernel<<<1, 32, 0, stream>>>();

  // weight prep
  cast_bf16_kernel<<<2048, 256, 0, stream>>>(wqkv, wqkv_b, (long)3 * HDIM * HDIM);
  cast_bf16_kernel<<<1024, 256, 0, stream>>>(wproj, wproj_b, (long)HDIM * HDIM);
  transpose_cast_kernel<<<dim3(FDIM / 32, HDIM / 32, EDIM), dim3(32, 8), 0, stream>>>(w1, w1t_b, HDIM, FDIM);
  transpose_cast_kernel<<<dim3(HDIM / 32, FDIM / 32, EDIM), dim3(32, 8), 0, stream>>>(w2, w2t_b, FDIM, HDIM);

  // attention block
  rmsnorm_cast_kernel<<<TDIM, 256, 0, stream>>>(hs, ln1w, xb);
  gemm_bf16_nt<<<dim3(3 * HDIM / 64, TDIM / 128), 256, 0, stream>>>(xb, wqkv_b, qkv_f, TDIM, 3 * HDIM, HDIM);
  rope_pack_kernel<<<dim3(SDIM, BDIM * NHD), 64, 0, stream>>>(qkv_f, qb, kbuf, vtb);
  flash_attn_kernel<<<dim3(SDIM / 64, BDIM * NHD), 128, 0, stream>>>(qb, kbuf, vtb, ctx_b);
  gemm_bf16_nt<<<dim3(HDIM / 64, TDIM / 128), 256, 0, stream>>>(ctx_b, wproj_b, hidden, TDIM, HDIM, HDIM);
  add_inplace_kernel<<<2048, 256, 0, stream>>>(hidden, hs, (long)TDIM * HDIM);

  // MoE block
  rmsnorm_cast_kernel<<<TDIM, 256, 0, stream>>>(hidden, ln2w, xb);
  router_logits_kernel<<<TDIM, 256, 0, stream>>>(xb, rw, logits);
  router_topk_kernel<<<(TDIM + 255) / 256, 256, 0, stream>>>(logits, probs);
  zero_kernel<<<2048, 256, 0, stream>>>(yacc, (long)TDIM * HDIM);
  for (int e = 0; e < EDIM; ++e) {
    gemm_bf16_nt<<<dim3(FDIM / 64, TDIM / 128), 256, 0, stream>>>(
        xb, w1t_b + (size_t)e * FDIM * HDIM, h1_f, TDIM, FDIM, HDIM);
    gelu_cast_kernel<<<4096, 256, 0, stream>>>(h1_f, h1g_b, (long)TDIM * FDIM);
    gemm_bf16_nt<<<dim3(HDIM / 64, TDIM / 128), 256, 0, stream>>>(
        h1g_b, w2t_b + (size_t)e * HDIM * FDIM, y_f, TDIM, HDIM, FDIM);
    combine_kernel<<<2048, 256, 0, stream>>>(yacc, y_f, probs, e, (long)TDIM * HDIM);
  }
  final_add_kernel<<<2048, 256, 0, stream>>>(yacc, hidden, out, (long)TDIM * HDIM);
}
